// early_fusion_67508295959191
// MI455X (gfx1250) — matrix operand order repaired and checked by reading, not compiled, not run on hardware
//
#include <hip/hip_runtime.h>
#include <hip/hip_bf16.h>

#define WHD 224
#define NPIX (WHD * WHD)        // 50176
#define BATCH 128
#define NCLS 400
#define KITERS (NPIX / 32)      // 1568 wmma K-steps of 32
#define WAVES 8                 // waves per block (256 threads, wave32)
#define KSPLIT 7                // grid.z K-splits; 7*8 = 56 slices, 1568/56 = 28 exact
#define SLICES (KSPLIT * WAVES) // 56

typedef __attribute__((ext_vector_type(16))) _Float16 v16h;
typedef __attribute__((ext_vector_type(8)))  _Float16 v8h;
typedef __attribute__((ext_vector_type(8)))  float    v8f;

// ---------------------------------------------------------------------------
// Kernel 1: per-(batch, frame) rectangle extraction (faithful mask logic).
// ---------------------------------------------------------------------------
__global__ void rects_kernel(const float* __restrict__ skel, int4* __restrict__ rects) {
  int tid = blockIdx.x * blockDim.x + threadIdx.x;
  if (tid >= BATCH * 15) return;
  int b = tid / 15, f = tid % 15;
  int src = (f < 8) ? f : f + 1;                 // skip mid frame (T/2 == 8)
  const float* sk  = skel + ((size_t)(b * 16 + src) * 25) * 3;
  const float* mid = skel + ((size_t)(b * 16 + 8)   * 25) * 3;

  float bestd = -1e30f; int bestj = 0;
  for (int j = 0; j < 25; ++j) {
    float x = sk[j*3+0], y = sk[j*3+1], z = sk[j*3+2];
    float dx = x - mid[j*3+0], dy = y - mid[j*3+1], dz = z - mid[j*3+2];
    float d = sqrtf(dx*dx + dy*dy + dz*dz);
    if (d > 1.0f) d = 0.0f;
    bool valid = (x >= 0.f) && (x <= 1.f) && (y >= 0.f) && (y <= 1.f);
    if (!valid) d = -1.0f;
    if (d > bestd) { bestd = d; bestj = j; }     // first-max, matches argmax
  }
  float selx = sk[bestj*3+0], sely = sk[bestj*3+1];
  int px = (int)(selx * WHD);
  int py = (int)(sely * WHD);
  const int S = 15;
  bool cxl = (px - S) < 0,  cxh = (px + S) > WHD;
  bool cyl = (py - S) < 0,  cyh = (py + S) > WHD;
  bool corner = (cxl || cxh) && (cyl || cyh);
  bool els = !corner;
  int r0 = (cxl && (cyl || cyh)) ? 0   : px - S;
  int r1 = (cxh && (cyl || cyh)) ? WHD : px + S;
  if (els && cxl) { r0 = 0; r1 = 0; }
  int c0 = ((cxl || cxh) && cyl) ? 0   : py - S;
  int c1 = ((cxl || cxh) && cyh) ? WHD : py + S;
  if (els && cyl) { c0 = 0; c1 = 0; }
  rects[tid] = make_int4(r0, r1, c0, c1);
}

// ---------------------------------------------------------------------------
// Kernel 2: rasterize union-of-rects into an f16 {0,1} mask image [B][NPIX].
// ---------------------------------------------------------------------------
__global__ void mask_kernel(const int4* __restrict__ rects, _Float16* __restrict__ Mh) {
  const int b = blockIdx.x;
  __shared__ int4 sr[15];
  if (threadIdx.x < 15) sr[threadIdx.x] = rects[b * 15 + threadIdx.x];
  __syncthreads();
  for (int g = threadIdx.x; g < NPIX / 8; g += blockDim.x) {
    int p  = g * 8;                 // 8 pixels, never crosses a row (224 % 8 == 0)
    int i  = p / WHD;
    int j0 = p % WHD;
    unsigned mbits = 0u;
    for (int f = 0; f < 15; ++f) {
      int4 rc = sr[f];
      if (i >= rc.x && i < rc.y) {
        int lo = rc.z - j0; if (lo < 0) lo = 0;
        int hi = rc.w - j0; if (hi > 8) hi = 8;
        for (int t = lo; t < hi; ++t) mbits |= (1u << t);
      }
    }
    v8h out;
#pragma unroll
    for (int t = 0; t < 8; ++t)
      out[t] = ((mbits >> t) & 1u) ? (_Float16)1.0f : (_Float16)0.0f;
    *(v8h*)(Mh + (size_t)b * NPIX + p) = out;
  }
}

// ---------------------------------------------------------------------------
// GEMM helpers: per-lane operand fragment loads (contiguous k order k = 16*half + s, same for A and B).
// ---------------------------------------------------------------------------
__device__ __forceinline__ v16h load_a_frag(const _Float16* __restrict__ p) {
  // A (16-bit 16x32): lane -> M = lane&15; hi-half lanes take K+16..31.
  v8h lo = *(const v8h*)(p);
  v8h hiv = *(const v8h*)(p + 8);
  v16h a;
#pragma unroll
  for (int i = 0; i < 8; ++i) { a[i] = lo[i]; a[i + 8] = hiv[i]; }
  return a;
}

__device__ __forceinline__ v16h load_b_frag_f32(const float* __restrict__ p) {
  // B (32x16): lane -> N = lane&15; hi-half lanes take K+16..31. cvt f32->f16.
  v8f b0 = *(const v8f*)(p);
  v8f b1 = *(const v8f*)(p + 8);
  v16h b;
#pragma unroll
  for (int i = 0; i < 8; ++i) { b[i] = (_Float16)b0[i]; b[i + 8] = (_Float16)b1[i]; }
  return b;
}

// ---------------------------------------------------------------------------
// Kernel 3: split-K WMMA GEMM, two M-tiles per wave sharing one B fragment.
//   partial[z][m][n] = sum over this z's K slices of mask[m,:] . W[n,:]
// K loop unrolled x2 (exact: 28 iters/wave) -> 4 v_wmma per loop body.
// ---------------------------------------------------------------------------
__global__ __launch_bounds__(256) void gemm_kernel(const _Float16* __restrict__ Mh,
                                                   const float* __restrict__ W,
                                                   float* __restrict__ partial) {
  const int lane = threadIdx.x & 31;
  const int wv   = threadIdx.x >> 5;
  const int nb   = blockIdx.x * 16;     // class tile (N)
  const int mb   = blockIdx.y * 32;     // paired batch tiles (M, M+16)
  const int r16  = lane & 15;
  const int hi   = lane >> 4;

  const _Float16* aptr0 = Mh + (size_t)(mb + r16) * NPIX + hi * 16;
  const _Float16* aptr1 = aptr0 + (size_t)16 * NPIX;
  const float*    bptr  = W  + (size_t)(nb + r16) * NPIX + hi * 16;

  v8f acc0 = {}, acc1 = {};
  const int slice = blockIdx.z * WAVES + wv;   // 0..55, each owns 28 K-steps
  for (int it = slice; it < KITERS; it += 2 * SLICES) {
    {
      const size_t k = (size_t)it * 32;
      __builtin_prefetch(bptr + k + 32 * 2 * SLICES, 0, 3);   // global_prefetch_b8
      v16h bm = load_b_frag_f32(bptr + k);
      v16h a0 = load_a_frag(aptr0 + k);
      v16h a1 = load_a_frag(aptr1 + k);
      acc0 = __builtin_amdgcn_wmma_f32_16x16x32_f16(false, a0, false, bm,
                                                    (short)0, acc0, false, false);
      acc1 = __builtin_amdgcn_wmma_f32_16x16x32_f16(false, a1, false, bm,
                                                    (short)0, acc1, false, false);
    }
    {
      const size_t k = (size_t)(it + SLICES) * 32;
      v16h bm = load_b_frag_f32(bptr + k);
      v16h a0 = load_a_frag(aptr0 + k);
      v16h a1 = load_a_frag(aptr1 + k);
      acc0 = __builtin_amdgcn_wmma_f32_16x16x32_f16(false, a0, false, bm,
                                                    (short)0, acc0, false, false);
      acc1 = __builtin_amdgcn_wmma_f32_16x16x32_f16(false, a1, false, bm,
                                                    (short)0, acc1, false, false);
    }
  }

  // Reduce the 8 K-slice waves of this block through LDS (both M tiles).
  __shared__ float red[WAVES][512];
#pragma unroll
  for (int r = 0; r < 8; ++r) {
    red[wv][r * 32 + lane]       = acc0[r];
    red[wv][256 + r * 32 + lane] = acc1[r];
  }
  __syncthreads();

#pragma unroll
  for (int half = 0; half < 2; ++half) {
    const int t = threadIdx.x + half * 256;
    float s = 0.f;
#pragma unroll
    for (int w2 = 0; w2 < WAVES; ++w2) s += red[w2][t];
    // C/D layout: VGPR r, lane l -> M = r + 8*(l>>4), N = l&15.
    const int rr = (t >> 5) & 7, l2 = t & 31, tile = t >> 8;
    const int m = mb + tile * 16 + rr + 8 * (l2 >> 4);
    const int n = nb + (l2 & 15);
    partial[((size_t)blockIdx.z * BATCH + m) * NCLS + n] = s;
  }
}

// ---------------------------------------------------------------------------
// Kernel 4: out = rgb * (sum_z partial + bias)   (fixed-order, deterministic)
// ---------------------------------------------------------------------------
__global__ void final_kernel(const float* __restrict__ rgb,
                             const float* __restrict__ partial,
                             const float* __restrict__ bias,
                             float* __restrict__ out) {
  int idx = blockIdx.x * blockDim.x + threadIdx.x;
  if (idx >= BATCH * NCLS) return;
  int n = idx % NCLS;
  float s = 0.f;
#pragma unroll
  for (int z = 0; z < KSPLIT; ++z) s += partial[(size_t)z * BATCH * NCLS + idx];
  out[idx] = rgb[idx] * (s + bias[n]);
}

// ---------------------------------------------------------------------------
extern "C" void kernel_launch(void* const* d_in, const int* in_sizes, int n_in,
                              void* d_out, int out_size, void* d_ws, size_t ws_size,
                              hipStream_t stream) {
  const float* rgb  = (const float*)d_in[0];   // [128,400]
  const float* skel = (const float*)d_in[1];   // [128,16,25,3]
  const float* W    = (const float*)d_in[2];   // [400,50176]
  const float* bias = (const float*)d_in[3];   // [400]
  float* out = (float*)d_out;                  // [128,400]

  char* ws = (char*)d_ws;
  float*    partial = (float*)ws;                         // 7*128*400*4 = 1,433,600 B
  int4*     rects   = (int4*)(ws + 1433600);              // 128*15*16   =    30,720 B
  _Float16* Mh      = (_Float16*)(ws + 1433600 + 30720);  // 128*50176*2 = 12,845,056 B

  rects_kernel<<<(BATCH * 15 + 255) / 256, 256, 0, stream>>>(skel, rects);
  mask_kernel<<<BATCH, 256, 0, stream>>>(rects, Mh);
  gemm_kernel<<<dim3(NCLS / 16, BATCH / 32, KSPLIT), 256, 0, stream>>>(Mh, W, partial);
  final_kernel<<<(BATCH * NCLS + 255) / 256, 256, 0, stream>>>(rgb, partial, bias, out);
}
